// SingleHeadSelfAttention_30468497997942
// MI455X (gfx1250) — matrix operand order repaired and checked by reading, not compiled, not run on hardware
//
#include <hip/hip_runtime.h>
#include <hip/hip_bf16.h>

typedef __attribute__((ext_vector_type(16))) __bf16 v16bf;
typedef __attribute__((ext_vector_type(8)))  __bf16 v8bf;
typedef __attribute__((ext_vector_type(8)))  float  v8f;
typedef __attribute__((ext_vector_type(4)))  int    v4i;

#define N_EMBD 1024
#define BATCH  4
#define SEQ    2048
#define KC     128           // K elements per LDS stage
#define BPITCH (KC + 8)      // padded LDS row pitch (272B = 17x16B, bank-conflict free)

#if __has_builtin(__builtin_amdgcn_global_load_async_to_lds_b128)
#define HAS_ASYNC_LDS 1
#else
#define HAS_ASYNC_LDS 0
#endif

typedef __attribute__((address_space(1))) v4i* gv4i_ptr;   // global int4*
typedef __attribute__((address_space(3))) v4i* lv4i_ptr;   // LDS int4*

// -------------------- helpers --------------------
__device__ __forceinline__ v8f wmma_bf16(v16bf a, v16bf b, v8f c) {
    return __builtin_amdgcn_wmma_f32_16x16x32_bf16(false, a, false, b,
                                                   (short)0, c, false, false);
}

// copy 64 contiguous bytes global -> LDS (per lane addresses; 4 async b128 ops)
__device__ __forceinline__ void stage_copy64(const __bf16* g, __bf16* l) {
#if HAS_ASYNC_LDS
    gv4i_ptr gp = (gv4i_ptr)(g);
    lv4i_ptr lp = (lv4i_ptr)(l);
    __builtin_amdgcn_global_load_async_to_lds_b128(gp, lp, 0,  0);
    __builtin_amdgcn_global_load_async_to_lds_b128(gp, lp, 16, 0);
    __builtin_amdgcn_global_load_async_to_lds_b128(gp, lp, 32, 0);
    __builtin_amdgcn_global_load_async_to_lds_b128(gp, lp, 48, 0);
#else
    uint4* ld = (uint4*)l;
    const uint4* gd = (const uint4*)g;
    ld[0] = gd[0]; ld[1] = gd[1]; ld[2] = gd[2]; ld[3] = gd[3];
#endif
}

__device__ __forceinline__ void wait_async_lds() {
#if HAS_ASYNC_LDS
#if __has_builtin(__builtin_amdgcn_s_wait_asynccnt)
    __builtin_amdgcn_s_wait_asynccnt(0);
#else
    asm volatile("s_wait_asynccnt 0x0" ::: "memory");
#endif
#endif
}

// -------------------- fp32 -> bf16 convert --------------------
__global__ __launch_bounds__(256)
void cvt_f32_bf16(const float* __restrict__ in, __bf16* __restrict__ out, int n) {
    int i = (blockIdx.x * 256 + threadIdx.x) * 4;
    if (i < n) {
        float4 v = *(const float4*)(in + i);
        out[i + 0] = (__bf16)v.x;
        out[i + 1] = (__bf16)v.y;
        out[i + 2] = (__bf16)v.z;
        out[i + 3] = (__bf16)v.w;
    }
}

// -------------------- generic NT GEMM: Out = A * B^T --------------------
// A: [M x K] row-major bf16, Bm: [N x K] row-major bf16, Out: [M x ldo] f32/bf16.
// Block tile 256M x 64N (8 waves, each 32M x 64N). B panel double-buffered in LDS
// via async-to-LDS; A streamed from global per wave.
// Requires: M % 256 == 0, N % 64 == 0, K % 128 == 0.
template<bool OUT_BF16>
__global__ __launch_bounds__(256)
void gemm_nt(const __bf16* __restrict__ A, int lda, long long aBatch,
             const __bf16* __restrict__ Bm, int ldb, long long bBatch,
             void* __restrict__ Out, int ldo, long long oBatch,
             int K, int causal)
{
    // causal skip: q rows [256*bx, 256*bx+255], k cols [64*by, 64*by+63]
    if (causal && (int)(blockIdx.y * 64) > (int)(blockIdx.x * 256) + 255) return;

    __shared__ __bf16 Bsm[2][64][BPITCH];

    const int tid  = threadIdx.x;
    const int lane = tid & 31;
    const int wave = tid >> 5;
    const int r    = lane & 15;
    const int hi   = lane >> 4;

    const __bf16* Ab = A  + (long long)blockIdx.z * aBatch;
    const __bf16* Bb = Bm + (long long)blockIdx.z * bBatch;

    const int mrow = blockIdx.x * 256 + wave * 32;
    const int ncol = blockIdx.y * 64;

    // cooperative B-panel staging: thread t handles 64B of row (t>>2), quad (t&3)
    const int crow = tid >> 2;
    const int cq   = (tid & 3) * 32;
    const __bf16* gB = Bb + (size_t)(ncol + crow) * ldb + cq;
    __bf16* lB[2] = { &Bsm[0][crow][cq], &Bsm[1][crow][cq] };

    // A frag pointers (16x32 bf16): lane holds row (lane&15),
    // K 0..15 contiguous (lo half) or 16..31 (hi half)
    const __bf16* ap0 = Ab + (size_t)(mrow + r) * lda + (hi << 4);
    const __bf16* ap1 = ap0 + (size_t)16 * lda;

    v8f acc[8] = {};

    const int nstages = K / KC;
    stage_copy64(gB, lB[0]);                    // prologue: stage 0 -> buf 0

    for (int s = 0; s < nstages; ++s) {
        wait_async_lds();                       // this wave's stage s complete
        __syncthreads();                        // all waves' stage s visible
        if (s + 1 < nstages)                    // overlap: stream stage s+1
            stage_copy64(gB + (size_t)(s + 1) * KC, lB[(s + 1) & 1]);

        const int kbase = s * KC;
        #pragma unroll
        for (int kk = 0; kk < KC; kk += 32) {
            union AU { v16bf v; v8bf h[2]; } a0, a1;
            const __bf16* a0p = ap0 + kbase + kk;
            const __bf16* a1p = ap1 + kbase + kk;
            a0.h[0] = *(const v8bf*)(a0p);
            a0.h[1] = *(const v8bf*)(a0p + 8);
            a1.h[0] = *(const v8bf*)(a1p);
            a1.h[1] = *(const v8bf*)(a1p + 8);
            __builtin_prefetch(a0p + 256, 0, 3);     // WGP-scope stream prefetch
            __builtin_prefetch(a1p + 256, 0, 3);

            #pragma unroll
            for (int nt = 0; nt < 4; ++nt) {
                // B frag (32x16): lane = col (lane&15), 16 contiguous K (+16 hi half)
                v16bf b = *(const v16bf*)&Bsm[s & 1][nt * 16 + r][kk + (hi << 4)];
                acc[nt]     = wmma_bf16(a0.v, b, acc[nt]);
                acc[4 + nt] = wmma_bf16(a1.v, b, acc[4 + nt]);
            }
        }
        // no trailing barrier needed: buffer s&1 is only overwritten by the
        // stage issued after the *next* barrier, which proves all reads finished
    }

    // C/D layout: VGPR j, lane l -> (M = j + 8*(l>>4), N = l&15)
    const int m0 = mrow + hi * 8;
    if constexpr (OUT_BF16) {
        __bf16* Ob = (__bf16*)Out + (long long)blockIdx.z * oBatch;
        #pragma unroll
        for (int g = 0; g < 2; ++g)
            #pragma unroll
            for (int nt = 0; nt < 4; ++nt)
                #pragma unroll
                for (int j = 0; j < 8; ++j)
                    Ob[(size_t)(m0 + g * 16 + j) * ldo + ncol + nt * 16 + r] =
                        (__bf16)acc[g * 4 + nt][j];
    } else {
        float* Ob = (float*)Out + (long long)blockIdx.z * oBatch;
        #pragma unroll
        for (int g = 0; g < 2; ++g)
            #pragma unroll
            for (int nt = 0; nt < 4; ++nt)
                #pragma unroll
                for (int j = 0; j < 8; ++j)
                    Ob[(size_t)(m0 + g * 16 + j) * ldo + ncol + nt * 16 + r] =
                        acc[g * 4 + nt][j];
    }
}

// -------------------- causal softmax, one wave32 per row --------------------
__global__ __launch_bounds__(256)
void softmax_causal(const float* __restrict__ S, __bf16* __restrict__ P) {
    const int lane = threadIdx.x & 31;
    const int wave = threadIdx.x >> 5;
    const int row  = blockIdx.x * 8 + wave;     // 0 .. B*SEQ-1
    const int qi   = row & (SEQ - 1);
    const float* s = S + (size_t)row * SEQ;
    __bf16*      p = P + (size_t)row * SEQ;
    const float scale = 0.03125f;               // 1/sqrt(1024)

    float m = -3.402823466e38f;
    for (int j = lane; j <= qi; j += 32) m = fmaxf(m, s[j]);
    #pragma unroll
    for (int o = 16; o; o >>= 1) m = fmaxf(m, __shfl_xor(m, o, 32));

    float sum = 0.0f;
    for (int j = lane; j <= qi; j += 32) sum += __expf((s[j] - m) * scale);
    #pragma unroll
    for (int o = 16; o; o >>= 1) sum += __shfl_xor(sum, o, 32);

    const float inv = 1.0f / sum;
    for (int j = lane; j < SEQ; j += 32)
        p[j] = (j <= qi) ? (__bf16)(__expf((s[j] - m) * scale) * inv) : (__bf16)0.0f;
}

// -------------------- launcher --------------------
extern "C" void kernel_launch(void* const* d_in, const int* in_sizes, int n_in,
                              void* d_out, int out_size, void* d_ws, size_t ws_size,
                              hipStream_t stream) {
    const float* x  = (const float*)d_in[0];
    const float* Wq = (const float*)d_in[1];
    const float* Wk = (const float*)d_in[2];
    const float* Wv = (const float*)d_in[3];
    float* out = (float*)d_out;

    char* ws = (char*)d_ws;
    __bf16* xb  = (__bf16*)(ws);                              // 16 MiB  [8192 x 1024]
    __bf16* wqb = (__bf16*)(ws + (16ll << 20));               //  2 MiB
    __bf16* wkb = wqb + (1 << 20);                            //  2 MiB
    __bf16* wvb = wkb + (1 << 20);                            //  2 MiB
    __bf16* qb  = (__bf16*)(ws + (22ll << 20));               // 16 MiB  [8192 x 1024]
    __bf16* kb  = (__bf16*)(ws + (38ll << 20));               // 16 MiB
    __bf16* vtb = (__bf16*)(ws + (54ll << 20));               // 16 MiB  V^T [1024 x 8192]
    float*  Sf  = (float*) (ws + (70ll << 20));               // 64 MiB  [4][2048 x 2048]
    __bf16* Pb  = (__bf16*)(ws + (134ll << 20));              // 32 MiB  [4][2048 x 2048]

    const int nX = BATCH * SEQ * N_EMBD;      // 8388608
    const int nW = N_EMBD * N_EMBD;           // 1048576

    cvt_f32_bf16<<<nX / 1024, 256, 0, stream>>>(x,  xb,  nX);
    cvt_f32_bf16<<<nW / 1024, 256, 0, stream>>>(Wq, wqb, nW);
    cvt_f32_bf16<<<nW / 1024, 256, 0, stream>>>(Wk, wkb, nW);
    cvt_f32_bf16<<<nW / 1024, 256, 0, stream>>>(Wv, wvb, nW);

    // q = x * Wq^T  [8192 x 1024]
    gemm_nt<true><<<dim3(32, 16, 1), 256, 0, stream>>>(
        xb, N_EMBD, 0, wqb, N_EMBD, 0, qb, N_EMBD, 0, N_EMBD, 0);
    // k = x * Wk^T
    gemm_nt<true><<<dim3(32, 16, 1), 256, 0, stream>>>(
        xb, N_EMBD, 0, wkb, N_EMBD, 0, kb, N_EMBD, 0, N_EMBD, 0);
    // V^T = Wv * x^T  [1024 x 8192]  (same NT form: A=Wv, B=x)
    gemm_nt<true><<<dim3(4, 128, 1), 256, 0, stream>>>(
        wvb, N_EMBD, 0, xb, N_EMBD, 0, vtb, BATCH * SEQ, 0, N_EMBD, 0);

    // S = q * k^T per batch, fp32, causal tile skip
    gemm_nt<false><<<dim3(8, 32, BATCH), 256, 0, stream>>>(
        qb, N_EMBD, (long long)SEQ * N_EMBD,
        kb, N_EMBD, (long long)SEQ * N_EMBD,
        Sf, SEQ,    (long long)SEQ * SEQ,
        N_EMBD, 1);

    // P = softmax(S / 32) with causal mask, bf16
    softmax_causal<<<BATCH * SEQ / 8, 256, 0, stream>>>(Sf, Pb);

    // out = P * V  (= P * (V^T)^T, NT form with B = V^T rows), fp32
    gemm_nt<false><<<dim3(8, 16, BATCH), 256, 0, stream>>>(
        Pb,  SEQ,         (long long)SEQ * SEQ,
        vtb, BATCH * SEQ, (long long)SEQ,
        out, N_EMBD,      (long long)SEQ * N_EMBD,
        SEQ, 0);
}
